// StandardMoELayer_87101936763282
// MI455X (gfx1250) — matrix operand order repaired and checked by reading, not compiled, not run on hardware
//
#include <hip/hip_runtime.h>

typedef __attribute__((ext_vector_type(16))) _Float16 f16x16;
typedef __attribute__((ext_vector_type(8)))  float    f32x8;
typedef __attribute__((ext_vector_type(4)))  unsigned int u32x4;
typedef __attribute__((ext_vector_type(8)))  int      i32x8;
typedef __attribute__((ext_vector_type(4)))  int      i32x4;

constexpr int B_  = 8, S_ = 4096, D_ = 512, DFF = 2048, NE = 8;
constexpr int T_  = B_ * S_;          // 32768 tokens
constexpr int KT1 = D_  / 32;         // 16 k-steps in GEMM1
constexpr int KT2 = DFF / 32;         // 64 k-steps in GEMM2
constexpr int ROWS = 32;              // M rows per block (2 x 16-row subtiles)
constexpr int TILE_CAP = 2056;        // ceil((2T + 8*31)/32)
constexpr int ENT_CAP  = TILE_CAP * ROWS;

// padded LDS strides (f16 elements): +4 dwords per row -> 4-bank rotation
constexpr int ASTR = D_  + 8;         // 520  (1040 B, 16B aligned)
constexpr int HSTR = DFF + 8;         // 2056 (4112 B, 16B aligned)
constexpr int LDS_A_BYTES = ROWS * ASTR * 2;            // 33280
constexpr int LDS_H_BYTES = ROWS * HSTR * 2;            // 131584
constexpr int LDS_BYTES   = LDS_A_BYTES + LDS_H_BYTES;  // 164864 (<320KB WGP)

// ---- workspace layout (32-bit word offsets) ----
constexpr int W_SUM    = 0;
constexpr int W_SUMSQ  = 8;
constexpr int W_COUNTS = 16;
constexpr int W_CURS   = 24;
constexpr int W_ETOK   = 32;
constexpr int W_EW     = W_ETOK + ENT_CAP;   // 65824
constexpr int W_MEAN   = W_EW + ENT_CAP;     // 131616
constexpr int W_RSTD   = W_MEAN + 8;
constexpr int W_TLOG   = W_RSTD + 8;
constexpr int W_POFF   = W_TLOG + 64;
constexpr int W_NT     = W_POFF + 9;
constexpr int W_TOPE   = 131712;
constexpr int W_TOPP   = W_TOPE + 2 * T_;
constexpr int W_XF16   = W_TOPP + 2 * T_;    // 262784 (byte ofs 16B aligned)
constexpr int W_W1P    = W_XF16 + (T_ * D_) / 2;
constexpr int W_W2P    = W_W1P + (NE * D_ * DFF) / 2;
constexpr int ZERO_WORDS = W_MEAN;

// ---- output layout ----
constexpr int OUT_LB   = T_ * D_;
constexpr int OUT_IDX  = OUT_LB + 1;
constexpr int OUT_TASK = OUT_IDX + 2 * T_;

static __device__ __forceinline__ unsigned short f2h(float f) {
  union { _Float16 h; unsigned short u; } c;
  c.h = (_Float16)f;
  return c.u;
}

union Frag { uint4 q[2]; f16x16 v; };

// ---------------- init kernels ----------------
__global__ __launch_bounds__(256) void zero_ws_kernel(int* wsi) {
  int i = blockIdx.x * 256 + threadIdx.x;
  if (i < ZERO_WORDS) wsi[i] = 0;
}

__global__ __launch_bounds__(256) void zero_out_kernel(float* out) {
  size_t i = (size_t)blockIdx.x * 256 + threadIdx.x;
  if (i < (size_t)T_ * D_) out[i] = 0.f;
}

// ---------------- batch LN statistics ----------------
__global__ __launch_bounds__(256)
void reduce_kernel(const float* __restrict__ x, float* sum, float* sumsq) {
  int b   = blockIdx.x >> 7;
  int blk = blockIdx.x & 127;
  const float* xb = x + (size_t)b * (S_ * D_) + (size_t)blk * 16384;
  float s = 0.f, q = 0.f;
  #pragma unroll 4
  for (int i = 0; i < 64; ++i) {
    float v = xb[threadIdx.x + (i << 8)];
    s += v; q += v * v;
  }
  #pragma unroll
  for (int off = 16; off; off >>= 1) {
    s += __shfl_xor(s, off, 32);
    q += __shfl_xor(q, off, 32);
  }
  __shared__ float ls[8], lq[8];
  int wave = threadIdx.x >> 5;
  if ((threadIdx.x & 31) == 0) { ls[wave] = s; lq[wave] = q; }
  __syncthreads();
  if (threadIdx.x == 0) {
    float S1 = 0.f, S2 = 0.f;
    for (int i = 0; i < 8; ++i) { S1 += ls[i]; S2 += lq[i]; }
    unsafeAtomicAdd(&sum[b], S1);
    unsafeAtomicAdd(&sumsq[b], S2);
  }
}

// ---------------- finalize stats + task logits + task_id output ----------------
__global__ void task_finalize_kernel(const float* sum, const float* sumsq,
                                     const int* task_id, const float* task_emb,
                                     const float* gate_w, const float* gate_b,
                                     float* meanArr, float* rstdArr, float* tlog,
                                     float* out) {
  int tid = threadIdx.x;
  if (tid < 8) {
    float inv = 1.f / (float)(S_ * D_);
    float m   = sum[tid] * inv;
    float var = sumsq[tid] * inv - m * m;
    meanArr[tid] = m;
    rstdArr[tid] = rsqrtf(var + 1e-5f);
    out[OUT_TASK + tid] = (float)task_id[tid];
  }
  int b = tid >> 3, e = tid & 7;
  int tk = task_id[b];
  float acc = gate_b[e];
  #pragma unroll 8
  for (int j = 0; j < 64; ++j)
    acc += task_emb[tk * 64 + j] * gate_w[(D_ + j) * NE + e];
  tlog[tid] = acc;
}

// ---------------- gating: one wave per token ----------------
__global__ __launch_bounds__(256)
void gating_kernel(const float* __restrict__ x, const float* __restrict__ gate_w,
                   const float* meanArr, const float* rstdArr, const float* tlog,
                   unsigned short* xh, int* topk_e, float* topk_p, int* counts,
                   float* out) {
  __shared__ float gwx[NE * D_];
  for (int idx = threadIdx.x; idx < NE * D_; idx += 256) {
    int e = idx >> 9, d = idx & (D_ - 1);
    gwx[idx] = gate_w[d * NE + e];
  }
  __syncthreads();
  int wave = threadIdx.x >> 5, lane = threadIdx.x & 31;
  int t = blockIdx.x * 8 + wave;
  int b = t >> 12;
  float m = meanArr[b], r = rstdArr[b];
  const float* xr = x + (size_t)t * D_;
  unsigned short* xo = xh + (size_t)t * D_;
  float acc[8] = {0.f, 0.f, 0.f, 0.f, 0.f, 0.f, 0.f, 0.f};
  #pragma unroll
  for (int i = 0; i < 16; ++i) {
    int d = lane + (i << 5);
    float xv = xr[d];
    xo[d] = f2h(xv);                  // experts consume RAW x (fp16)
    float xn = (xv - m) * r;          // gate consumes x_norm
    #pragma unroll
    for (int e = 0; e < 8; ++e) acc[e] += xn * gwx[e * D_ + d];
  }
  #pragma unroll
  for (int e = 0; e < 8; ++e)
    #pragma unroll
    for (int off = 16; off; off >>= 1) acc[e] += __shfl_xor(acc[e], off, 32);
  if (lane == 0) {
    float lg[8];
    #pragma unroll
    for (int e = 0; e < 8; ++e) lg[e] = acc[e] + tlog[b * 8 + e];
    int e0 = 0;
    for (int e = 1; e < 8; ++e) if (lg[e] > lg[e0]) e0 = e;
    int e1 = (e0 == 0) ? 1 : 0;
    for (int e = 0; e < 8; ++e) if (e != e0 && lg[e] > lg[e1]) e1 = e;
    float p0 = 1.f / (1.f + __expf(lg[e1] - lg[e0]));
    float p1 = 1.f - p0;
    topk_e[t * 2] = e0; topk_e[t * 2 + 1] = e1;
    topk_p[t * 2] = p0; topk_p[t * 2 + 1] = p1;
    atomicAdd(&counts[e0], 1);
    atomicAdd(&counts[e1], 1);
    out[OUT_IDX + t * 2]     = (float)e0;
    out[OUT_IDX + t * 2 + 1] = (float)e1;
  }
}

// ---------------- prefix offsets (padded to 32) + load-balance loss ----------------
__global__ void offsets_lb_kernel(const int* counts, int* cursors, int* poff,
                                  int* ntile, float* out) {
  int off = 0;
  poff[0] = 0;
  for (int e = 0; e < NE; ++e) {
    cursors[e] = off;
    off += ((counts[e] + 31) >> 5) << 5;   // pad each expert to 32-row tiles
    poff[e + 1] = off;
  }
  ntile[0] = off >> 5;
  float mu = 0.f;
  for (int e = 0; e < NE; ++e) mu += (float)counts[e];
  mu *= (1.f / NE);
  float ss = 0.f;
  for (int e = 0; e < NE; ++e) { float d = (float)counts[e] - mu; ss += d * d; }
  float um = mu + 1e-6f;
  out[OUT_LB] = (ss / (NE - 1)) / (um * um);
}

// ---------------- scatter tokens into per-expert entry lists ----------------
__global__ __launch_bounds__(256)
void scatter_kernel(const int* topk_e, const float* topk_p, int* cursors,
                    int* etok, float* ew) {
  int t = blockIdx.x * 256 + threadIdx.x;
  #pragma unroll
  for (int k = 0; k < 2; ++k) {
    int e = topk_e[t * 2 + k];
    int pos = atomicAdd(&cursors[e], 1);
    etok[pos] = t;
    ew[pos] = topk_p[t * 2 + k];
  }
}

// ---------------- pack weights into WMMA-B-fragment layout ----------------
__global__ __launch_bounds__(256)
void pack_weights_kernel(const float* __restrict__ w1, const float* __restrict__ w2,
                         unsigned short* w1p, unsigned short* w2p) {
  int idx = blockIdx.x * 256 + threadIdx.x;
  constexpr int N1 = NE * D_ * DFF;
  if (idx < N1) {
    int kk = idx & 31; int r = idx >> 5;
    int n = r & (DFF - 1); r >>= 11;
    int kt = r & 15; int e = r >> 4;
    w1p[idx] = f2h(w1[((size_t)(e * D_ + kt * 32 + kk)) * DFF + n]);
  } else {
    int j = idx - N1;
    int kk = j & 31; int r = j >> 5;
    int n = r & (D_ - 1); r >>= 9;
    int kt = r & 63; int e = r >> 6;
    w2p[j] = f2h(w2[((size_t)(e * DFF + kt * 32 + kk)) * D_ + n]);
  }
}

// ---------------- fused expert FFN (32-row tiles) ----------------
__global__ __launch_bounds__(512)
void moe_kernel(const unsigned short* __restrict__ xh,
                const unsigned short* __restrict__ w1p,
                const unsigned short* __restrict__ w2p,
                const float* __restrict__ b1, const float* __restrict__ b2,
                const int* __restrict__ etok, const float* __restrict__ ew,
                const int* __restrict__ poff, const int* __restrict__ ntile,
                float* __restrict__ out) {
  extern __shared__ unsigned short smem[];
  unsigned short* Abuf = smem;                       // [32][ASTR] f16
  unsigned short* H    = smem + ROWS * ASTR;         // [32][HSTR] f16

  int tile = blockIdx.x;
  if (tile >= ntile[0]) return;
  int row0 = tile << 5;
  int e = 0;
  #pragma unroll
  for (int i = 1; i <= NE; ++i) if (row0 >= poff[i]) e = i;

  const int lane = threadIdx.x & 31;
  const int wave = threadIdx.x >> 5;   // 16 waves
  const bool lo  = lane < 16;
  const int  lm  = lane & 15;
  const int  c0  = lo ? 0 : 8;         // A-frag K-chunk base (ISA layout)
  const int  bb  = lo ? 0 : 8;         // B-frag kk base (packed layout)

  // ---- stage 32 gathered token rows into LDS via TDM gather mode ----
#if __has_builtin(__builtin_amdgcn_tensor_load_to_lds) && __has_builtin(__builtin_amdgcn_s_wait_tensorcnt)
  if (wave == 0) {
    unsigned long long gaddr = (unsigned long long)(uintptr_t)xh;
    unsigned lds0 = (unsigned)(uintptr_t)Abuf;   // low 32 bits = LDS byte offset
    #pragma unroll
    for (int s2 = 0; s2 < 2; ++s2) {
      u32x4 g0;
      g0[0] = 0x80000001u;                       // count=1, gather_mode=1, 16b idx
      g0[1] = lds0 + (unsigned)(s2 * 16 * ASTR * 2);
      g0[2] = (unsigned)(gaddr & 0xffffffffu);
      g0[3] = (unsigned)((gaddr >> 32) & 0x01ffffffu) | 0x80000000u; // type=2
      i32x8 g1;
      // data_size=2B; pad_enable, pad every 256 dwords (code 7) by 4 dwords (code 3)
      g1[0] = (1 << 16) | (1 << 20) | (7 << 22) | (3 << 25);
      g1[1] = (int)((unsigned)(D_ & 0xffff) << 16);                // tensor_dim0 lo
      g1[2] = (int)(((unsigned)D_ >> 16) | ((unsigned)(T_ & 0xffff) << 16));
      g1[3] = (int)(((unsigned)T_ >> 16) | ((unsigned)D_ << 16));  // tile_dim0=512
      g1[4] = 16;                                // tile_dim1 = #row indices
      g1[5] = D_;                                // tensor_dim0_stride
      g1[6] = 0; g1[7] = 0;
      i32x4 g2, g3;
      #pragma unroll
      for (int jj = 0; jj < 4; ++jj) {
        int bi = row0 + s2 * 16 + jj * 2;
        g2[jj] = (etok[bi] & 0xffff) | (etok[bi + 1] << 16);
        g3[jj] = (etok[bi + 8] & 0xffff) | (etok[bi + 9] << 16);
      }
      i32x8 g4;                                  // extra operand (6-arg form): zero
      #pragma unroll
      for (int jj = 0; jj < 8; ++jj) g4[jj] = 0;
      __builtin_amdgcn_tensor_load_to_lds(g0, g1, g2, g3, g4, 0);
    }
    __builtin_amdgcn_s_wait_tensorcnt(0);
  }
#else
  for (int i = threadIdx.x; i < ROWS * (D_ / 8); i += 512) {
    int rrow = i >> 6;                 // 64 uint4 per row
    int col  = i & 63;
    int tok  = etok[row0 + rrow];
    *(uint4*)(Abuf + rrow * ASTR + col * 8) =
        ((const uint4*)(xh + (size_t)tok * D_))[col];
  }
#endif
  __syncthreads();

  // ---- GEMM1: [32 x 512] @ [512 x 2048]; 8 N-tiles/wave, B reused for 2 M-subs ----
  f32x8 acc[2][8];
  #pragma unroll
  for (int s = 0; s < 2; ++s)
    #pragma unroll
    for (int i = 0; i < 8; ++i)
      #pragma unroll
      for (int r = 0; r < 8; ++r) acc[s][i][r] = 0.f;

  const unsigned short* w1e = w1p + (size_t)e * (KT1 * DFF * 32);
  const unsigned short* a0  = Abuf + lm * ASTR + c0;
  const unsigned short* a1  = Abuf + (16 + lm) * ASTR + c0;

  for (int kt = 0; kt < KT1; ++kt) {
    Frag A0, A1;
    A0.q[0] = *(const uint4*)(a0 + kt * 32);
    A0.q[1] = *(const uint4*)(a0 + kt * 32 + 16);
    A1.q[0] = *(const uint4*)(a1 + kt * 32);
    A1.q[1] = *(const uint4*)(a1 + kt * 32 + 16);
    const unsigned short* bbase = w1e + ((size_t)kt * DFF) * 32 + bb;
    #pragma unroll
    for (int i = 0; i < 8; ++i) {
      int n = ((wave << 3) + i) * 16 + lm;
      const unsigned short* bp = bbase + (size_t)n * 32;
      Frag Bf;
      Bf.q[0] = *(const uint4*)(bp);
      Bf.q[1] = *(const uint4*)(bp + 16);
      acc[0][i] = __builtin_amdgcn_wmma_f32_16x16x32_f16(
          false, A0.v, false, Bf.v, (short)0, acc[0][i], false, false);
      acc[1][i] = __builtin_amdgcn_wmma_f32_16x16x32_f16(
          false, A1.v, false, Bf.v, (short)0, acc[1][i], false, false);
    }
  }

  // bias + SiLU -> H (f16 in LDS)
  #pragma unroll
  for (int i = 0; i < 8; ++i) {
    int n = ((wave << 3) + i) * 16 + lm;
    float bias = b1[e * DFF + n];
    #pragma unroll
    for (int s = 0; s < 2; ++s)
      #pragma unroll
      for (int r = 0; r < 8; ++r) {
        int Mr = s * 16 + r + (lo ? 0 : 8);
        float v = acc[s][i][r] + bias;
        v = v / (1.f + __expf(-v));
        H[Mr * HSTR + n] = f2h(v);
      }
  }
  __syncthreads();

  // ---- GEMM2: [32 x 2048] @ [2048 x 512]; 2 N-tiles/wave, B reused for 2 M-subs ----
  f32x8 acc2[2][2];
  #pragma unroll
  for (int s = 0; s < 2; ++s)
    #pragma unroll
    for (int i = 0; i < 2; ++i)
      #pragma unroll
      for (int r = 0; r < 8; ++r) acc2[s][i][r] = 0.f;

  const unsigned short* w2e = w2p + (size_t)e * (KT2 * D_ * 32);
  const unsigned short* h0  = H + lm * HSTR + c0;
  const unsigned short* h1  = H + (16 + lm) * HSTR + c0;

  for (int kt = 0; kt < KT2; ++kt) {
    Frag A0, A1;
    A0.q[0] = *(const uint4*)(h0 + kt * 32);
    A0.q[1] = *(const uint4*)(h0 + kt * 32 + 16);
    A1.q[0] = *(const uint4*)(h1 + kt * 32);
    A1.q[1] = *(const uint4*)(h1 + kt * 32 + 16);
    const unsigned short* bbase = w2e + ((size_t)kt * D_) * 32 + bb;
    #pragma unroll
    for (int i = 0; i < 2; ++i) {
      int n = ((wave << 1) + i) * 16 + lm;
      const unsigned short* bp = bbase + (size_t)n * 32;
      Frag Bf;
      Bf.q[0] = *(const uint4*)(bp);
      Bf.q[1] = *(const uint4*)(bp + 16);
      acc2[0][i] = __builtin_amdgcn_wmma_f32_16x16x32_f16(
          false, A0.v, false, Bf.v, (short)0, acc2[0][i], false, false);
      acc2[1][i] = __builtin_amdgcn_wmma_f32_16x16x32_f16(
          false, A1.v, false, Bf.v, (short)0, acc2[1][i], false, false);
    }
  }

  // bias, gate-prob scale, accumulate into output (f32 global atomic)
  #pragma unroll
  for (int i = 0; i < 2; ++i) {
    int n = ((wave << 1) + i) * 16 + lm;
    float bias = b2[e * D_ + n];
    #pragma unroll
    for (int s = 0; s < 2; ++s)
      #pragma unroll
      for (int r = 0; r < 8; ++r) {
        int Mr  = s * 16 + r + (lo ? 0 : 8);
        int tok = etok[row0 + Mr];
        float wgt = ew[row0 + Mr];           // 0 for pad rows -> adds 0
        float v = (acc2[s][i][r] + bias) * wgt;
        unsafeAtomicAdd(&out[(size_t)tok * D_ + n], v);
      }
  }
}

// ---------------- host-side launcher ----------------
extern "C" void kernel_launch(void* const* d_in, const int* in_sizes, int n_in,
                              void* d_out, int out_size, void* d_ws, size_t ws_size,
                              hipStream_t stream) {
  const float* x        = (const float*)d_in[0];
  const int*   task_id  = (const int*)d_in[1];
  const float* w1       = (const float*)d_in[2];
  const float* b1       = (const float*)d_in[3];
  const float* w2       = (const float*)d_in[4];
  const float* b2       = (const float*)d_in[5];
  const float* task_emb = (const float*)d_in[6];
  const float* gate_w   = (const float*)d_in[7];
  const float* gate_b   = (const float*)d_in[8];
  float* out = (float*)d_out;

  int*   wsi = (int*)d_ws;
  float* wsf = (float*)d_ws;
  float* sum     = wsf + W_SUM;
  float* sumsq   = wsf + W_SUMSQ;
  int*   counts  = wsi + W_COUNTS;
  int*   cursors = wsi + W_CURS;
  int*   etok    = wsi + W_ETOK;
  float* ewt     = wsf + W_EW;
  float* meanA   = wsf + W_MEAN;
  float* rstdA   = wsf + W_RSTD;
  float* tlog    = wsf + W_TLOG;
  int*   poff    = wsi + W_POFF;
  int*   ntile   = wsi + W_NT;
  int*   tope    = wsi + W_TOPE;
  float* topp    = wsf + W_TOPP;
  unsigned short* xh  = (unsigned short*)(wsi + W_XF16);
  unsigned short* w1p = (unsigned short*)(wsi + W_W1P);
  unsigned short* w2p = (unsigned short*)(wsi + W_W2P);

  zero_ws_kernel<<<(ZERO_WORDS + 255) / 256, 256, 0, stream>>>(wsi);
  zero_out_kernel<<<(T_ * D_) / 256, 256, 0, stream>>>(out);
  pack_weights_kernel<<<(2 * NE * D_ * DFF) / 256, 256, 0, stream>>>(w1, w2, w1p, w2p);
  reduce_kernel<<<B_ * 128, 256, 0, stream>>>(x, sum, sumsq);
  task_finalize_kernel<<<1, 64, 0, stream>>>(sum, sumsq, task_id, task_emb,
                                             gate_w, gate_b, meanA, rstdA, tlog, out);
  gating_kernel<<<T_ / 8, 256, 0, stream>>>(x, gate_w, meanA, rstdA, tlog,
                                            xh, tope, topp, counts, out);
  offsets_lb_kernel<<<1, 1, 0, stream>>>(counts, cursors, poff, ntile, out);
  scatter_kernel<<<T_ / 256, 256, 0, stream>>>(tope, topp, cursors, etok, ewt);
  moe_kernel<<<TILE_CAP, 512, LDS_BYTES, stream>>>(xh, w1p, w2p, b1, b2, etok, ewt,
                                                   poff, ntile, out);
}
